// SOMNetwork_81020263072059
// MI455X (gfx1250) — matrix operand order repaired and checked by reading, not compiled, not run on hardware
//
#include <hip/hip_runtime.h>
#include <hip/hip_bf16.h>

typedef __attribute__((ext_vector_type(16))) _Float16 v16h;
typedef __attribute__((ext_vector_type(8)))  _Float16 v8h;
typedef __attribute__((ext_vector_type(8)))  float    v8f;

union frag16 { v16h v; v8h h[2]; };

// ---------------------------------------------------------------------------
// Epilogue for one 16x16 D tile: d = max(x2 + w2 - 2*acc, 0); triangle; crelu.
// NHWC flat store is just M*COUT + N (no spatial decode needed).
// D layout: lane L holds col N=L%16, vgpr r -> row M = r + 8*(L>>4).
// x2own: lane L holds x2 of row (L&15); broadcast via bpermute.
// ---------------------------------------------------------------------------
template <int COUT>
__device__ __forceinline__ void rbf_epi(
    const v8f& acc, float x2own, float w2n, float tri, float bias,
    int Mbase, int g, int N, _Float16* __restrict__ out)
{
  const float tinv = 1.0f / tri;
#pragma unroll
  for (int r = 0; r < 8; ++r) {
    const int row = r + 8 * g;
    const float x2 = __shfl(x2own, row, 32);
    float d = fmaxf(x2 + w2n - 2.0f * acc[r], 0.0f);
    d = fminf(d, tri);
    float t = 1.0f - d * tinv;
    t = (t >= bias) ? t : 0.0f;
    out[(size_t)(Mbase + row) * COUT + N] = (_Float16)t;
  }
}

// ---------------------------------------------------------------------------
// Vectorized RBF conv (CIN % 8 == 0, KTOT % 32 == 0), NHWC activations.
// Each wave computes a 32(M) x 32(N) macro-tile: 4 WMMAs per k-step sharing
// 2 A-fragments and 2 B-fragments (2 b128 loads per WMMA). K-order permuted
// to K' = (ky*KS+kx)*CIN + ci so each A-octet is one contiguous b128 load;
// OOB octets read a zero scratch line via address select (no exec branches).
// ---------------------------------------------------------------------------
template <int CIN, int COUT, int HIN, int WIN, int KS, int PAD, int HOUT, int WOUT>
__global__ __launch_bounds__(128) void rbf_conv_wmma_v(
    const _Float16* __restrict__ in,      // [256,HIN,WIN,CIN] f16
    const _Float16* __restrict__ wfrag,   // packed B fragments
    const float*    __restrict__ w2,      // [COUT]
    const float*    __restrict__ x2map,   // [256*HOUT*WOUT]
    const _Float16* __restrict__ zero16,  // 16B of zeros
    _Float16*       __restrict__ out,     // [256,HOUT,WOUT,COUT] f16
    float tri, float bias)
{
  constexpr int KTOT   = CIN * KS * KS;   // multiple of 32
  constexpr int KSTEPS = KTOT / 32;
  constexpr int NT2    = COUT / 32;       // n-tile pairs
  constexpr int HW     = HOUT * WOUT;

  const int tile = blockIdx.x * 4 + threadIdx.y;   // over (Mtiles/2)*NT2
  const int mt2 = tile / NT2, np = tile % NT2;
  const int lane = threadIdx.x;
  const int g = lane >> 4, lr = lane & 15;

  const int M0 = mt2 * 32 + lr;                    // rows of A sub-tile 0
  const int M1 = M0 + 16;                          // rows of A sub-tile 1
  const int b0 = M0 / HW, sp0 = M0 % HW, oy0 = sp0 / WOUT, ox0 = sp0 % WOUT;
  const int b1 = M1 / HW, sp1 = M1 % HW, oy1 = sp1 / WOUT, ox1 = sp1 % WOUT;
  const int n0 = np * 32 + lr;                     // cols of B sub-tile 0

  const _Float16* wf0 = wfrag + ((size_t)(np * 2) * KSTEPS * 32 + lane) * 16;
  const _Float16* wf1 = wf0 + (size_t)KSTEPS * 512;

  v8f acc00 = {}, acc01 = {}, acc10 = {}, acc11 = {};
  for (int ks = 0; ks < KSTEPS; ++ks) {
    if (ks + 1 < KSTEPS) {                         // global_prefetch_b8 path
      __builtin_prefetch(wf0 + (size_t)(ks + 1) * 512, 0, 1);
      __builtin_prefetch(wf1 + (size_t)(ks + 1) * 512, 0, 1);
    }
    frag16 a0, a1, bf0, bf1;
#pragma unroll
    for (int o = 0; o < 2; ++o) {
      const int kb  = ks * 32 + o * 16 + 8 * g;    // octet K' base
      const int pos = kb / CIN;                    // ky*KS + kx
      const int ci  = kb % CIN;                    // multiple of 8
      const int ky = pos / KS, kx = pos % KS;
      {
        const int iy = oy0 + ky - PAD, ix = ox0 + kx - PAD;
        const bool inb = (unsigned)iy < (unsigned)HIN && (unsigned)ix < (unsigned)WIN;
        const _Float16* p =
            inb ? in + (((size_t)(b0 * HIN + iy) * WIN + ix) * CIN + ci) : zero16;
        a0.h[o] = *(const v8h*)p;
      }
      {
        const int iy = oy1 + ky - PAD, ix = ox1 + kx - PAD;
        const bool inb = (unsigned)iy < (unsigned)HIN && (unsigned)ix < (unsigned)WIN;
        const _Float16* p =
            inb ? in + (((size_t)(b1 * HIN + iy) * WIN + ix) * CIN + ci) : zero16;
        a1.h[o] = *(const v8h*)p;
      }
    }
    bf0.h[0] = *(const v8h*)(wf0 + (size_t)ks * 512);
    bf0.h[1] = *(const v8h*)(wf0 + (size_t)ks * 512 + 8);
    bf1.h[0] = *(const v8h*)(wf1 + (size_t)ks * 512);
    bf1.h[1] = *(const v8h*)(wf1 + (size_t)ks * 512 + 8);
    acc00 = __builtin_amdgcn_wmma_f32_16x16x32_f16(false, a0.v, false, bf0.v, (short)0, acc00, false, false);
    acc01 = __builtin_amdgcn_wmma_f32_16x16x32_f16(false, a0.v, false, bf1.v, (short)0, acc01, false, false);
    acc10 = __builtin_amdgcn_wmma_f32_16x16x32_f16(false, a1.v, false, bf0.v, (short)0, acc10, false, false);
    acc11 = __builtin_amdgcn_wmma_f32_16x16x32_f16(false, a1.v, false, bf1.v, (short)0, acc11, false, false);
  }
  const float x20 = x2map[M0], x21 = x2map[M1];
  const float w20 = w2[n0],    w21 = w2[n0 + 16];
  rbf_epi<COUT>(acc00, x20, w20, tri, bias, mt2 * 32,      g, n0,      out);
  rbf_epi<COUT>(acc01, x20, w21, tri, bias, mt2 * 32,      g, n0 + 16, out);
  rbf_epi<COUT>(acc10, x21, w20, tri, bias, mt2 * 32 + 16, g, n0,      out);
  rbf_epi<COUT>(acc11, x21, w21, tri, bias, mt2 * 32 + 16, g, n0 + 16, out);
}

// ---------------------------------------------------------------------------
// Layer-0 specialization: CIN=1, KTOT=25 (single k-step), scalar A gather
// with zero-pointer trick; COUT=32 so one wave covers the whole N extent.
// ---------------------------------------------------------------------------
__global__ __launch_bounds__(128) void rbf_conv_wmma_l0(
    const _Float16* __restrict__ in,      // [256,26,26] f16
    const _Float16* __restrict__ wfrag,   // [2][32][16] halves
    const float* __restrict__ w2, const float* __restrict__ x2map,
    const _Float16* __restrict__ zero16, _Float16* __restrict__ out,
    float tri, float bias)
{
  constexpr int HIN = 26, WIN = 26, KS = 5, PAD = 3, HOUT = 28, WOUT = 28;
  constexpr int COUT = 32, KTOT = 25, HW = HOUT * WOUT;
  const int mt2 = blockIdx.x * 4 + threadIdx.y;    // Mtiles/2 = 6272
  const int lane = threadIdx.x;
  const int g = lane >> 4, lr = lane & 15;
  const int M0 = mt2 * 32 + lr, M1 = M0 + 16;
  const int b0 = M0 / HW, sp0 = M0 % HW, oy0 = sp0 / WOUT, ox0 = sp0 % WOUT;
  const int b1 = M1 / HW, sp1 = M1 % HW, oy1 = sp1 / WOUT, ox1 = sp1 % WOUT;

  frag16 a0, a1, bf0, bf1;
#pragma unroll
  for (int e = 0; e < 16; ++e) {
    const int K  = (e < 8) ? (e + 8 * g) : (e + 8 + 8 * g);
    const int ky = K / KS, kx = K % KS;
    {
      const int iy = oy0 + ky - PAD, ix = ox0 + kx - PAD;
      const bool ok = (K < KTOT) && (unsigned)iy < (unsigned)HIN && (unsigned)ix < (unsigned)WIN;
      a0.v[e] = *(ok ? in + ((size_t)b0 * HIN + iy) * WIN + ix : zero16);
    }
    {
      const int iy = oy1 + ky - PAD, ix = ox1 + kx - PAD;
      const bool ok = (K < KTOT) && (unsigned)iy < (unsigned)HIN && (unsigned)ix < (unsigned)WIN;
      a1.v[e] = *(ok ? in + ((size_t)b1 * HIN + iy) * WIN + ix : zero16);
    }
  }
  bf0.h[0] = *(const v8h*)(wfrag + (size_t)lane * 16);
  bf0.h[1] = *(const v8h*)(wfrag + (size_t)lane * 16 + 8);
  bf1.h[0] = *(const v8h*)(wfrag + (size_t)(32 + lane) * 16);
  bf1.h[1] = *(const v8h*)(wfrag + (size_t)(32 + lane) * 16 + 8);
  v8f acc00 = {}, acc01 = {}, acc10 = {}, acc11 = {};
  acc00 = __builtin_amdgcn_wmma_f32_16x16x32_f16(false, a0.v, false, bf0.v, (short)0, acc00, false, false);
  acc01 = __builtin_amdgcn_wmma_f32_16x16x32_f16(false, a0.v, false, bf1.v, (short)0, acc01, false, false);
  acc10 = __builtin_amdgcn_wmma_f32_16x16x32_f16(false, a1.v, false, bf0.v, (short)0, acc10, false, false);
  acc11 = __builtin_amdgcn_wmma_f32_16x16x32_f16(false, a1.v, false, bf1.v, (short)0, acc11, false, false);
  const float x20 = x2map[M0], x21 = x2map[M1];
  const float w20 = w2[lr], w21 = w2[lr + 16];
  rbf_epi<COUT>(acc00, x20, w20, tri, bias, mt2 * 32,      g, lr,      out);
  rbf_epi<COUT>(acc01, x20, w21, tri, bias, mt2 * 32,      g, lr + 16, out);
  rbf_epi<COUT>(acc10, x21, w20, tri, bias, mt2 * 32 + 16, g, lr,      out);
  rbf_epi<COUT>(acc11, x21, w21, tri, bias, mt2 * 32 + 16, g, lr + 16, out);
}

// ---------------------------------------------------------------------------
// Prep kernels (tiny, run once per call).
// ---------------------------------------------------------------------------
__global__ void pack_wfrag(const float* __restrict__ w, _Float16* __restrict__ wfrag,
                           int CIN, int KS, int KTOT, int KSTEPS, int total)
{
  int idx = blockIdx.x * blockDim.x + threadIdx.x;
  if (idx >= total) return;
  const int e    = idx & 15;
  const int lane = (idx >> 4) & 31;
  const int ks   = (idx >> 9) % KSTEPS;
  const int nt   = idx / (KSTEPS * 512);
  const int N  = nt * 16 + (lane & 15);
  const int Kp = ks * 32 + (e & 7) + 8 * (lane >> 4) + 16 * (e >> 3);
  float v = 0.0f;
  if (Kp < KTOT) {
    const int K = (CIN > 1) ? ((Kp % CIN) * KS * KS + Kp / CIN) : Kp;
    v = w[N * KTOT + K];
  }
  wfrag[idx] = (_Float16)v;
}

__global__ void calc_w2(const float* __restrict__ w, float* __restrict__ w2,
                        int COUT, int KTOT)
{
  int n = blockIdx.x * blockDim.x + threadIdx.x;
  if (n >= COUT) return;
  float s = 0.0f;
  for (int k = 0; k < KTOT; ++k) {
    const float v = (float)(_Float16)w[n * KTOT + k];
    s = fmaf(v, v, s);
  }
  w2[n] = s;
}

__global__ void chansum(const _Float16* __restrict__ in, float* __restrict__ s,
                        int total, int C)
{
  int idx = blockIdx.x * blockDim.x + threadIdx.x;
  if (idx >= total) return;
  const _Float16* p = in + (size_t)idx * C;
  float acc = 0.0f;
  for (int c = 0; c < C; ++c) {
    const float v = (float)p[c];
    acc = fmaf(v, v, acc);
  }
  s[idx] = acc;
}

__global__ void winsum(const float* __restrict__ s, float* __restrict__ x2map,
                       int HIN, int WIN, int HOUT, int WOUT, int KS, int PAD,
                       int total)
{
  int idx = blockIdx.x * blockDim.x + threadIdx.x;
  if (idx >= total) return;
  const int ox = idx % WOUT; int t = idx / WOUT;
  const int oy = t % HOUT;   const int b = t / HOUT;
  float acc = 0.0f;
  for (int ky = 0; ky < KS; ++ky) {
    const int iy = oy + ky - PAD;
    if ((unsigned)iy >= (unsigned)HIN) continue;
    for (int kx = 0; kx < KS; ++kx) {
      const int ix = ox + kx - PAD;
      if ((unsigned)ix < (unsigned)WIN)
        acc += s[((size_t)b * HIN + iy) * WIN + ix];
    }
  }
  x2map[idx] = acc;
}

template <int C, int H, int W>
__global__ void sfm_nhwc(const _Float16* __restrict__ in,
                         _Float16* __restrict__ out, float alpha)
{
  constexpr int HO = H / 2, WO = W / 2, CV = C / 8;
  constexpr int TOT = 256 * HO * WO * CV;
  int idx = blockIdx.x * blockDim.x + threadIdx.x;
  if (idx >= TOT) return;
  const int cv = idx % CV; int t = idx / CV;
  const int xo = t % WO;   t /= WO;
  const int yo = t % HO;   const int b = t / HO;
  const _Float16* p = in + ((size_t)(b * H + 2 * yo) * W + 2 * xo) * C + cv * 8;
  const v8h x00 = *(const v8h*)p;
  const v8h x01 = *(const v8h*)(p + C);
  const v8h x10 = *(const v8h*)(p + (size_t)W * C);
  const v8h x11 = *(const v8h*)(p + (size_t)W * C + C);
  const float a1 = alpha, a2 = alpha * alpha, a3 = a2 * alpha;
  v8h r;
#pragma unroll
  for (int j = 0; j < 8; ++j)
    r[j] = (_Float16)(((float)x00[j] * a3 + (float)x01[j] * a2 +
                       (float)x10[j] * a1 + (float)x11[j]) * 0.25f);
  *(v8h*)(out + (size_t)idx * 8) = r;
}

__global__ void extract_ch(const float* __restrict__ x,
                           _Float16* __restrict__ out, int ch)
{
  int idx = blockIdx.x * blockDim.x + threadIdx.x;
  if (idx >= 256 * 676) return;
  const int b = idx / 676, s = idx % 676;
  out[idx] = (_Float16)x[(b * 3 + ch) * 676 + s];
}

__global__ void zero_init(_Float16* z)
{
  if (threadIdx.x < 32) z[threadIdx.x] = (_Float16)0.0f;
}

// ---------------------------------------------------------------------------
// FC: [256,12288] x [12288,10] + b. Features are the three NHWC [256,16,256]
// branch buffers (j' = s*256 + c); weights pre-packed with matching K order.
// Each wave computes 32 rows (2 M-tiles) sharing the B fragments.
// ---------------------------------------------------------------------------
__global__ void pack_fcfrag(const float* __restrict__ fcw,
                            _Float16* __restrict__ frag)
{
  int idx = blockIdx.x * blockDim.x + threadIdx.x;
  if (idx >= 384 * 512) return;
  const int e    = idx & 15;
  const int lane = (idx >> 4) & 31;
  const int ks   = idx >> 9;
  const int N = lane & 15;
  float v = 0.0f;
  if (N < 10) {
    const int Kpp = ks * 32 + (e & 7) + 8 * (lane >> 4) + 16 * (e >> 3);
    const int i = Kpp >> 12;
    const int j = Kpp & 4095;
    const int s = j >> 8, c = j & 255;
    v = fcw[N * 12288 + i * 4096 + c * 16 + s];   // reference j = c*16 + s
  }
  frag[idx] = (_Float16)v;
}

__global__ __launch_bounds__(128) void fc_wmma(
    const _Float16* __restrict__ f0, const _Float16* __restrict__ f1,
    const _Float16* __restrict__ f2, const _Float16* __restrict__ frag,
    const float* __restrict__ fcb, float* __restrict__ out)
{
  const int mt2 = blockIdx.x * 4 + threadIdx.y;   // 8 macro-tiles (32 rows)
  const int lane = threadIdx.x;
  const int g = lane >> 4, lr = lane & 15;
  const int r0 = mt2 * 32 + lr;                   // batch rows

  v8f acc0 = {}, acc1 = {};
  for (int ks = 0; ks < 384; ++ks) {
    const _Float16* __restrict__ src = (ks < 128) ? f0 : ((ks < 256) ? f1 : f2);
    const int jb = (ks * 32) & 4095;
    frag16 a0, a1, bm;
    a0.h[0] = *(const v8h*)(src + (size_t)r0 * 4096 + jb + 8 * g);
    a0.h[1] = *(const v8h*)(src + (size_t)r0 * 4096 + jb + 16 + 8 * g);
    a1.h[0] = *(const v8h*)(src + (size_t)(r0 + 16) * 4096 + jb + 8 * g);
    a1.h[1] = *(const v8h*)(src + (size_t)(r0 + 16) * 4096 + jb + 16 + 8 * g);
    bm.h[0] = *(const v8h*)(frag + ((size_t)ks * 32 + lane) * 16);
    bm.h[1] = *(const v8h*)(frag + ((size_t)ks * 32 + lane) * 16 + 8);
    acc0 = __builtin_amdgcn_wmma_f32_16x16x32_f16(false, a0.v, false, bm.v, (short)0, acc0, false, false);
    acc1 = __builtin_amdgcn_wmma_f32_16x16x32_f16(false, a1.v, false, bm.v, (short)0, acc1, false, false);
  }
  if (lr < 10) {
    const float bb = fcb[lr];
#pragma unroll
    for (int r = 0; r < 8; ++r) {
      out[(mt2 * 32 + r + 8 * g) * 10 + lr]      = acc0[r] + bb;
      out[(mt2 * 32 + 16 + r + 8 * g) * 10 + lr] = acc1[r] + bb;
    }
  }
}

// ---------------------------------------------------------------------------
extern "C" void kernel_launch(void* const* d_in, const int* in_sizes, int n_in,
                              void* d_out, int out_size, void* d_ws, size_t ws_size,
                              hipStream_t stream)
{
  (void)out_size; (void)ws_size;
  const float* x = nullptr;
  const float* conv[3][4] = {};
  const float* fcw = nullptr;
  const float* fcb = nullptr;
  int cc[4] = {0, 0, 0, 0};
  for (int i = 0; i < n_in; ++i) {
    const int s = in_sizes[i];
    const float* p = (const float*)d_in[i];
    if      (s == 256 * 3 * 26 * 26) x = p;
    else if (s == 32 * 25   && cc[0] < 3) conv[cc[0]++][0] = p;
    else if (s == 64 * 288  && cc[1] < 3) conv[cc[1]++][1] = p;
    else if (s == 128 * 576 && cc[2] < 3) conv[cc[2]++][2] = p;
    else if (s == 256 * 128 && cc[3] < 3) conv[cc[3]++][3] = p;
    else if (s == 10 * 12288) fcw = p;
    else if (s == 10)         fcb = p;
  }
  if (!x || !fcw || !fcb || cc[0] < 3 || cc[1] < 3 || cc[2] < 3 || cc[3] < 3)
    return;

  char* wsb = (char*)d_ws;
  size_t cur = 0;
  auto carve = [&](size_t bytes) -> char* {
    char* p = wsb + cur;
    cur += (bytes + 255) & ~(size_t)255;
    return p;
  };
  _Float16* zero16 = (_Float16*)carve(64);
  _Float16* bufA   = (_Float16*)carve(13107200);            // pre-pool acts
  _Float16* bufB   = (_Float16*)carve(3276800);             // conv inputs
  _Float16* fin[3];
  for (int i = 0; i < 3; ++i) fin[i] = (_Float16*)carve(2097152);
  float* smap  = (float*)carve(173056 * 4);
  float* x2map = (float*)carve(200704 * 4);
  float* w2b[3][4];
  _Float16* wf[3][4];
  _Float16* fcfrag = (_Float16*)carve(384 * 512 * 2);
  static const int WCNT[4]  = {1024, 18432, 73728, 32768};
  static const int COUTL[4] = {32, 64, 128, 256};
  static const int CINL[4]  = {1, 32, 64, 128};
  static const int KSL[4]   = {5, 3, 3, 1};
  static const int KTOTL[4] = {25, 288, 576, 128};
  static const int KSTL[4]  = {1, 9, 18, 4};
  for (int br = 0; br < 3; ++br)
    for (int l = 0; l < 4; ++l) {
      wf[br][l]  = (_Float16*)carve((size_t)WCNT[l] * 2);
      w2b[br][l] = (float*)carve((size_t)COUTL[l] * 4);
    }

  const float TRI = 4.0f, ALPHA = 0.9f;
  const float BIAS[4] = {0.4f, 0.1f, 0.01f, 0.01f};
  dim3 cblk(32, 4);

  zero_init<<<1, 32, 0, stream>>>(zero16);
  for (int br = 0; br < 3; ++br)
    for (int l = 0; l < 4; ++l) {
      pack_wfrag<<<(WCNT[l] + 255) / 256, 256, 0, stream>>>(
          conv[br][l], wf[br][l], CINL[l], KSL[l], KTOTL[l], KSTL[l], WCNT[l]);
      calc_w2<<<(COUTL[l] + 63) / 64, 64, 0, stream>>>(
          conv[br][l], w2b[br][l], COUTL[l], KTOTL[l]);
    }
  pack_fcfrag<<<(384 * 512 + 255) / 256, 256, 0, stream>>>(fcw, fcfrag);

  for (int br = 0; br < 3; ++br) {
    extract_ch<<<(256 * 676 + 255) / 256, 256, 0, stream>>>(x, bufB, br);

    // block0: 26x26 -> conv5 pad3 -> 28x28 -> sfm -> 14x14
    chansum<<<(256 * 676 + 255) / 256, 256, 0, stream>>>(bufB, smap, 256 * 676, 1);
    winsum<<<(256 * 784 + 255) / 256, 256, 0, stream>>>(smap, x2map, 26, 26, 28, 28, 5, 3, 256 * 784);
    rbf_conv_wmma_l0<<<6272 / 4, cblk, 0, stream>>>(
        bufB, wf[br][0], w2b[br][0], x2map, zero16, bufA, TRI, BIAS[0]);
    sfm_nhwc<32, 28, 28><<<(256 * 196 * 4 + 255) / 256, 256, 0, stream>>>(bufA, bufB, ALPHA);

    // block1: 14x14 -> conv3 pad4 -> 20x20 -> sfm -> 10x10
    chansum<<<(256 * 196 + 255) / 256, 256, 0, stream>>>(bufB, smap, 256 * 196, 32);
    winsum<<<(256 * 400 + 255) / 256, 256, 0, stream>>>(smap, x2map, 14, 14, 20, 20, 3, 4, 256 * 400);
    rbf_conv_wmma_v<32, 64, 14, 14, 3, 4, 20, 20><<<6400 / 4, cblk, 0, stream>>>(
        bufB, wf[br][1], w2b[br][1], x2map, zero16, bufA, TRI, BIAS[1]);
    sfm_nhwc<64, 20, 20><<<(256 * 100 * 8 + 255) / 256, 256, 0, stream>>>(bufA, bufB, ALPHA);

    // block2: 10x10 -> conv3 pad0 -> 8x8 -> sfm -> 4x4
    chansum<<<(256 * 100 + 255) / 256, 256, 0, stream>>>(bufB, smap, 256 * 100, 64);
    winsum<<<(256 * 64 + 255) / 256, 256, 0, stream>>>(smap, x2map, 10, 10, 8, 8, 3, 0, 256 * 64);
    rbf_conv_wmma_v<64, 128, 10, 10, 3, 0, 8, 8><<<2048 / 4, cblk, 0, stream>>>(
        bufB, wf[br][2], w2b[br][2], x2map, zero16, bufA, TRI, BIAS[2]);
    sfm_nhwc<128, 8, 8><<<(256 * 16 * 16 + 255) / 256, 256, 0, stream>>>(bufA, bufB, ALPHA);

    // block3: 1x1 conv, no sfm -> persistent branch output (NHWC [256,16,256])
    chansum<<<(256 * 16 + 255) / 256, 256, 0, stream>>>(bufB, smap, 256 * 16, 128);
    winsum<<<(256 * 16 + 255) / 256, 256, 0, stream>>>(smap, x2map, 4, 4, 4, 4, 1, 0, 256 * 16);
    rbf_conv_wmma_v<128, 256, 4, 4, 1, 0, 4, 4><<<1024 / 4, cblk, 0, stream>>>(
        bufB, wf[br][3], w2b[br][3], x2map, zero16, fin[br], TRI, BIAS[3]);
  }

  fc_wmma<<<2, cblk, 0, stream>>>(fin[0], fin[1], fin[2], fcfrag, fcb, (float*)d_out);
}
